// ConnectTransformerLayer_62139586839039
// MI455X (gfx1250) — compile-verified, matrix operand order checked by reading, not run on hardware
//
#include <hip/hip_runtime.h>

// ---------------------------------------------------------------------------
// CDNA5 (gfx1250) implementation: wave32, v_wmma_f32_16x16x32_bf16 everywhere.
// Round 2: vectorized (b128) fragment loads, pre-packed weight fragments,
// transposed-V attention operand, hoisted epilogue branches, prefetch hints.
// ---------------------------------------------------------------------------

typedef __attribute__((ext_vector_type(16))) __bf16 v16bf;
typedef __attribute__((ext_vector_type(8)))  __bf16 v8bf;
typedef __attribute__((ext_vector_type(8)))  float  v8f;

#define NDPTS 12288
#define NFEAT 64
#define NBATCH 8
#define EPSBN 1e-4f

__constant__ int c_start[NBATCH] = {0, 1200, 2600, 4100, 5636, 7236, 8936, 10736};
__constant__ int c_count[NBATCH] = {1200, 1400, 1500, 1536, 1600, 1700, 1800, 1552};

__device__ __forceinline__ unsigned short f2bf(float f) {
  unsigned int u = __builtin_bit_cast(unsigned int, f);
  unsigned int r = u + 0x7fffu + ((u >> 16) & 1u);   // round-to-nearest-even
  return (unsigned short)(r >> 16);
}
__device__ __forceinline__ __bf16 bfbits(unsigned short u) {
  return __builtin_bit_cast(__bf16, u);
}

__device__ __forceinline__ v8f wmma_bf16(v16bf a, v16bf b, v8f c) {
  return __builtin_amdgcn_wmma_f32_16x16x32_bf16(false, a, false, b, (short)0, c,
                                                 false, false);
}

__device__ __forceinline__ v16bf cat8(v8bf lo, v8bf hi) {
  return __builtin_shufflevector(lo, hi, 0, 1, 2, 3, 4, 5, 6, 7,
                                 8, 9, 10, 11, 12, 13, 14, 15);
}

// A-fragment (16x32 bf16) from a row pointer: halves are contiguous 16B runs.
// lane<16: kb=0, lanes>=16: kb=8; elements {kb..kb+7, kb+16..kb+23}. (ISA 7.12.2)
__device__ __forceinline__ v16bf load_frag_rowptr(const unsigned short* rowptr,
                                                  int kofs, int lane) {
  const int kb = ((lane >> 4) ? 8 : 0) + kofs;
  const unsigned short* p = rowptr + kb;
  return cat8(*(const v8bf*)p, *(const v8bf*)(p + 16));
}

// A-fragment with per-wave relative rows (stride between rows given).
__device__ __forceinline__ v16bf load_frag_rowmajor(const unsigned short* base,
                                                    int stride, int lane) {
  return load_frag_rowptr(base + (lane & 15) * stride, 0, lane);
}

// B-fragment from pre-packed weight storage: lane*16 contiguous bf16 (32B).
__device__ __forceinline__ v16bf load_frag_packed(const unsigned short* base,
                                                  int lane) {
  const unsigned short* p = base + lane * 16;
  return cat8(*(const v8bf*)p, *(const v8bf*)(p + 8));
}

// B-fragment of V^T rows [k0,k0+32) (full chunk, contiguous in K).
__device__ __forceinline__ v16bf load_frag_vt(const unsigned short* VT, int k0,
                                              int ncol0, int lane) {
  const int col = ncol0 + (lane & 15);
  const int kb  = (lane >> 4) ? 8 : 0;
  const unsigned short* p = VT + col * NDPTS + k0 + kb;
  return cat8(*(const v8bf*)p, *(const v8bf*)(p + 16));
}

// Same with per-element clamp for the ragged tail chunk (stage-2 only).
__device__ __forceinline__ v16bf load_frag_vt_clamped(const unsigned short* VT,
                                                      int k0, int klim, int ncol0,
                                                      int lane) {
  const int col = ncol0 + (lane & 15);
  const int kb  = (lane >> 4) ? 8 : 0;
  const unsigned short* p = VT + col * NDPTS;
  v16bf f;
#pragma unroll
  for (int j = 0; j < 8; ++j) f[j] = bfbits(p[min(k0 + kb + j, klim - 1)]);
#pragma unroll
  for (int j = 0; j < 8; ++j) f[8 + j] = bfbits(p[min(k0 + kb + 16 + j, klim - 1)]);
  return f;
}

// ---------------------------------------------------------------------------
// f32 -> bf16 convert
// ---------------------------------------------------------------------------
__global__ void cvt_bf16_kernel(const float* __restrict__ X,
                                unsigned short* __restrict__ Y, int n) {
  int i = blockIdx.x * blockDim.x + threadIdx.x;
  if (i < n) Y[i] = f2bf(X[i]);
}

// ---------------------------------------------------------------------------
// Pack weights [taps][64][64] f32 (k-major) into WMMA B-fragment order bf16:
// idx = ((t*4 + nt)*2 + kc)*512 + lane*16 + j  -> contiguous per-lane 32B.
// ---------------------------------------------------------------------------
__global__ void pack_w_kernel(const float* __restrict__ W,
                              unsigned short* __restrict__ P, int taps) {
  int i = blockIdx.x * blockDim.x + threadIdx.x;
  if (i >= taps * 4096) return;
  int j    = i & 15;
  int lane = (i >> 4) & 31;
  int kc   = (i >> 9) & 1;
  int nt   = (i >> 10) & 3;
  int t    = i >> 12;
  int col  = lane & 15;
  int kb   = (lane >> 4) ? 8 : 0;
  int jj   = (j < 8) ? j : (j + 8);          // {0..7} then {16..23}
  int k    = kc * 32 + kb + jj;
  P[i] = f2bf(W[t * 4096 + k * NFEAT + nt * 16 + col]);
}

// ---------------------------------------------------------------------------
// C[M,64] = A[M,64] @ Wp (pre-packed 64x64). Optional f32 / bf16 / bf16-transposed.
// ---------------------------------------------------------------------------
__global__ void gemm64_kernel(const unsigned short* __restrict__ A,
                              const unsigned short* __restrict__ Wp,
                              float* __restrict__ Cf,
                              unsigned short* __restrict__ Cb,
                              unsigned short* __restrict__ CbT, int M) {
  const int lane = threadIdx.x & 31;
  const int wave = threadIdx.x >> 5;
  const int row0 = blockIdx.x * 64 + wave * 16;
  if (row0 >= M) return;
  v16bf a0 = load_frag_rowmajor(A + row0 * NFEAT, NFEAT, lane);
  v16bf a1 = load_frag_rowmajor(A + row0 * NFEAT + 32, NFEAT, lane);
  v8f acc[4];
#pragma unroll
  for (int nt = 0; nt < 4; ++nt) {
    v16bf b0 = load_frag_packed(Wp + (nt * 2 + 0) * 512, lane);
    v16bf b1 = load_frag_packed(Wp + (nt * 2 + 1) * 512, lane);
    v8f c = {};
    c = wmma_bf16(a0, b0, c);
    acc[nt] = wmma_bf16(a1, b1, c);
  }
  const int n  = lane & 15;
  const int mo = (lane >> 4) * 8;
  if (Cf) {
#pragma unroll
    for (int nt = 0; nt < 4; ++nt)
#pragma unroll
      for (int r = 0; r < 8; ++r)
        Cf[(row0 + mo + r) * NFEAT + nt * 16 + n] = acc[nt][r];
  }
  if (Cb) {
#pragma unroll
    for (int nt = 0; nt < 4; ++nt)
#pragma unroll
      for (int r = 0; r < 8; ++r)
        Cb[(row0 + mo + r) * NFEAT + nt * 16 + n] = f2bf(acc[nt][r]);
  }
  if (CbT) {  // transposed layout [64][M] for attention V operand
#pragma unroll
    for (int nt = 0; nt < 4; ++nt)
#pragma unroll
      for (int r = 0; r < 8; ++r)
        CbT[(nt * 16 + n) * M + row0 + mo + r] = f2bf(acc[nt][r]);
  }
}

// ---------------------------------------------------------------------------
// Flash attention core: one block = 64 query rows (4 waves x 16), streaming
// keys in chunks of 32, online softmax, P re-layout through LDS.
// V is consumed transposed ([64][NDPTS]).
// ---------------------------------------------------------------------------
__device__ void attn_block(const unsigned short* __restrict__ Q,
                           const unsigned short* __restrict__ K,
                           const unsigned short* __restrict__ VT,
                           unsigned short* __restrict__ Ob,
                           int qfirst, int qlim, int kfirst, int klim,
                           unsigned short* Pw /* per-wave 16x32 LDS */) {
  const int lane    = threadIdx.x & 31;
  const int wave    = threadIdx.x >> 5;
  const int rowbase = qfirst + wave * 16;
  const int arow    = min(rowbase + (lane & 15), qlim - 1);

  v16bf qa0 = load_frag_rowptr(Q + arow * NFEAT, 0, lane);
  v16bf qa1 = load_frag_rowptr(Q + arow * NFEAT, 32, lane);

  v8f o[4] = {};
  float mrow[8], lrow[8];
#pragma unroll
  for (int r = 0; r < 8; ++r) { mrow[r] = -3.0e38f; lrow[r] = 0.0f; }

  const int nkc = (klim - kfirst + 31) >> 5;
  for (int it = 0; it < nkc; ++it) {
    const int k0 = kfirst + it * 32;
    // prefetch next key chunk into cache while this chunk computes
    if (it + 1 < nkc) {
      int kn = min(k0 + 32 + (lane & 15), klim - 1);
      __builtin_prefetch(K + kn * NFEAT, 0, 1);
    }
    v8f s[2];
    // ---- S = Q @ K^T (two 16x16 tiles; K^T B-fragment == K A-fragment) ----
#pragma unroll
    for (int t = 0; t < 2; ++t) {
      const int krow = min(k0 + t * 16 + (lane & 15), klim - 1);
      v16bf kb0 = load_frag_rowptr(K + krow * NFEAT, 0, lane);
      v16bf kb1 = load_frag_rowptr(K + krow * NFEAT, 32, lane);
      v8f a = {};
      a = wmma_bf16(qa0, kb0, a);
      a = wmma_bf16(qa1, kb1, a);
      if (k0 + t * 16 + (lane & 15) >= klim) {
#pragma unroll
        for (int r = 0; r < 8; ++r) a[r] = -1.0e30f;
      }
      s[t] = a;
    }
    // ---- online softmax: row max (half-wave reduction, width 16) ----
    float scale[8];
#pragma unroll
    for (int r = 0; r < 8; ++r) {
      float v = fmaxf(s[0][r], s[1][r]);
#pragma unroll
      for (int off = 1; off < 16; off <<= 1) v = fmaxf(v, __shfl_xor(v, off, 16));
      float mnew = fmaxf(mrow[r], v);
      scale[r]   = __expf(mrow[r] - mnew);
      mrow[r]    = mnew;
    }
    // ---- P = exp(S - m) staged into per-wave LDS as a 16x32 bf16 A tile ----
#pragma unroll
    for (int t = 0; t < 2; ++t) {
      const int col = t * 16 + (lane & 15);
#pragma unroll
      for (int r = 0; r < 8; ++r) {
        float p = __expf(s[t][r] - mrow[r]);
        s[t][r] = p;
        Pw[((lane >> 4) * 8 + r) * 32 + col] = f2bf(p);
      }
    }
    // ---- row sums + rescale accumulators ----
#pragma unroll
    for (int r = 0; r < 8; ++r) {
      float v = s[0][r] + s[1][r];
#pragma unroll
      for (int off = 1; off < 16; off <<= 1) v += __shfl_xor(v, off, 16);
      lrow[r] = lrow[r] * scale[r] + v;
      o[0][r] *= scale[r]; o[1][r] *= scale[r];
      o[2][r] *= scale[r]; o[3][r] *= scale[r];
    }
    // LDS writes must land before the A-fragment readback (intra-wave xchg).
    asm volatile("s_wait_dscnt 0" ::: "memory");
    v16bf pa = load_frag_rowmajor(Pw, 32, lane);
    // ---- O += P @ V ----
    if (k0 + 32 <= klim) {
#pragma unroll
      for (int nt = 0; nt < 4; ++nt)
        o[nt] = wmma_bf16(pa, load_frag_vt(VT, k0, nt * 16, lane), o[nt]);
    } else {
#pragma unroll
      for (int nt = 0; nt < 4; ++nt)
        o[nt] = wmma_bf16(pa, load_frag_vt_clamped(VT, k0, klim, nt * 16, lane),
                          o[nt]);
    }
  }
  // ---- epilogue: O / l ----
  const int n  = lane & 15;
  const int mo = (lane >> 4) * 8;
#pragma unroll
  for (int r = 0; r < 8; ++r) {
    int m = rowbase + mo + r;
    if (m < qlim) {
      float inv = 1.0f / lrow[r];
#pragma unroll
      for (int nt = 0; nt < 4; ++nt)
        Ob[m * NFEAT + nt * 16 + n] = f2bf(o[nt][r] * inv);
    }
  }
}

__global__ void attn_stage1_kernel(const unsigned short* Q, const unsigned short* K,
                                   const unsigned short* VT, unsigned short* Ob) {
  __shared__ unsigned short P[4 * 16 * 32];
  attn_block(Q, K, VT, Ob, blockIdx.x * 64, NDPTS, 0, NDPTS,
             P + (threadIdx.x >> 5) * (16 * 32));
}

__global__ void attn_stage2_kernel(const unsigned short* Q, const unsigned short* K,
                                   const unsigned short* VT, unsigned short* Ob) {
  __shared__ unsigned short P[4 * 16 * 32];
  int blk = blockIdx.x, b = 0;
  for (;;) {
    int nt = (c_count[b] + 63) >> 6;
    if (blk < nt) break;
    blk -= nt; ++b;
  }
  const int s = c_start[b], e = c_start[b] + c_count[b];
  attn_block(Q, K, VT, Ob, s + blk * 64, e, s, e,
             P + (threadIdx.x >> 5) * (16 * 32));
}

// ---------------------------------------------------------------------------
// Rulebook sparse conv: Out[i] = sum_t F[nbr[i,t]] @ W[t] (nbr<0 -> zero row).
// Weights pre-packed into B-fragment order; straight-line, no barriers.
// ---------------------------------------------------------------------------
__global__ void gconv_kernel(const unsigned short* __restrict__ F,
                             const int* __restrict__ nbr, int taps,
                             const unsigned short* __restrict__ Wp,
                             float* __restrict__ Of,
                             unsigned short* __restrict__ Ob) {
  const int lane = threadIdx.x & 31;
  const int wave = threadIdx.x >> 5;
  const int row0 = blockIdx.x * 64 + wave * 16;
  const int arow = row0 + (lane & 15);
  v8f acc[4] = {};
  for (int t = 0; t < taps; ++t) {
    int j = nbr[arow * taps + t];
    v16bf a0 = {}, a1 = {};
    if (j >= 0) {
      a0 = load_frag_rowptr(F + j * NFEAT, 0, lane);
      a1 = load_frag_rowptr(F + j * NFEAT, 32, lane);
    }
    const unsigned short* wt = Wp + t * 4096;
#pragma unroll
    for (int nt = 0; nt < 4; ++nt) {
      acc[nt] = wmma_bf16(a0, load_frag_packed(wt + (nt * 2 + 0) * 512, lane),
                          acc[nt]);
      acc[nt] = wmma_bf16(a1, load_frag_packed(wt + (nt * 2 + 1) * 512, lane),
                          acc[nt]);
    }
  }
  const int n  = lane & 15;
  const int mo = (lane >> 4) * 8;
  if (Of) {
#pragma unroll
    for (int nt = 0; nt < 4; ++nt)
#pragma unroll
      for (int r = 0; r < 8; ++r)
        Of[(row0 + mo + r) * NFEAT + nt * 16 + n] = acc[nt][r];
  }
  if (Ob) {
#pragma unroll
    for (int nt = 0; nt < 4; ++nt)
#pragma unroll
      for (int r = 0; r < 8; ++r)
        Ob[(row0 + mo + r) * NFEAT + nt * 16 + n] = f2bf(acc[nt][r]);
  }
}

// ---------------------------------------------------------------------------
// BatchNorm stats: one block per feature column; mean + E[x^2]-mean^2.
// ---------------------------------------------------------------------------
__global__ void bn_stats_kernel(const float* __restrict__ X,
                                const float* __restrict__ Xadd, int N,
                                float* __restrict__ mv) {
  __shared__ float s1[256], s2[256];
  const int c = blockIdx.x;
  float a = 0.f, b = 0.f;
  if (Xadd) {
    for (int i = threadIdx.x; i < N; i += blockDim.x) {
      float v = X[i * NFEAT + c] + Xadd[i * NFEAT + c];
      a += v; b += v * v;
    }
  } else {
    for (int i = threadIdx.x; i < N; i += blockDim.x) {
      float v = X[i * NFEAT + c];
      a += v; b += v * v;
    }
  }
  s1[threadIdx.x] = a; s2[threadIdx.x] = b;
  __syncthreads();
  for (int s = 128; s > 0; s >>= 1) {
    if ((int)threadIdx.x < s) {
      s1[threadIdx.x] += s1[threadIdx.x + s];
      s2[threadIdx.x] += s2[threadIdx.x + s];
    }
    __syncthreads();
  }
  if (threadIdx.x == 0) {
    float mean = s1[0] / (float)N;
    mv[c]         = mean;
    mv[NFEAT + c] = s2[0] / (float)N - mean * mean;
  }
}

// y = bn(X (+Xadd)) (+Res) (relu?); writes f32 and/or bf16.
__global__ void bn_apply_kernel(const float* __restrict__ X,
                                const float* __restrict__ Xadd,
                                const float* __restrict__ Res,
                                const float* __restrict__ mv,
                                const float* __restrict__ gamma,
                                const float* __restrict__ beta, int relu,
                                float* __restrict__ Of,
                                unsigned short* __restrict__ Ob, int total) {
  int i = blockIdx.x * blockDim.x + threadIdx.x;
  if (i >= total) return;
  int c = i & (NFEAT - 1);
  float v = X[i];
  if (Xadd) v += Xadd[i];
  float y = (v - mv[c]) * rsqrtf(mv[NFEAT + c] + EPSBN) * gamma[c] + beta[c];
  if (Res) y += Res[i];
  if (relu) y = fmaxf(y, 0.f);
  if (Of) Of[i] = y;
  if (Ob) Ob[i] = f2bf(y);
}

// ---------------------------------------------------------------------------
// Host orchestration
// ---------------------------------------------------------------------------
extern "C" void kernel_launch(void* const* d_in, const int* in_sizes, int n_in,
                              void* d_out, int out_size, void* d_ws, size_t ws_size,
                              hipStream_t stream) {
  (void)in_sizes; (void)n_in; (void)out_size; (void)ws_size;
  const float* x_decoder = (const float*)d_in[0];
  const float* x_encoder = (const float*)d_in[1];
  const float* Wp1 = (const float*)d_in[2];
  const float* Wq  = (const float*)d_in[3];
  const float* Wk  = (const float*)d_in[4];
  const float* Wv  = (const float*)d_in[5];
  const float* Wt  = (const float*)d_in[6];
  const float* Wq1 = (const float*)d_in[7];
  const float* Wk1 = (const float*)d_in[8];
  const float* Wv1 = (const float*)d_in[9];
  const float* Wdown = (const float*)d_in[10];
  const float* W3t = (const float*)d_in[11];
  const float* W3a = (const float*)d_in[12];
  const float* W3b = (const float*)d_in[13];
  const float* bn_g = (const float*)d_in[14];
  const float* bn_b = (const float*)d_in[15];
  const int* nbr    = (const int*)d_in[16];
  const int* kv_nbr = (const int*)d_in[17];
  float* out = (float*)d_out;

  char* ws = (char*)d_ws;
  size_t off = 0;
  auto alloc = [&](size_t bytes) -> void* {
    void* p = ws + off;
    off = (off + bytes + 255) & ~(size_t)255;
    return p;
  };
  const int NE = NDPTS * NFEAT;  // 786432
  auto bfbuf  = [&](int n) { return (unsigned short*)alloc((size_t)n * 2); };
  auto f32buf = [&](int n) { return (float*)alloc((size_t)n * 4); };

  // bf16 inputs + packed weights
  unsigned short* Xd16 = bfbuf(NE);
  unsigned short* Xe16 = bfbuf(NE);
  unsigned short* wp1_p = bfbuf(4096);
  unsigned short* wq_p  = bfbuf(4096);
  unsigned short* wk_p  = bfbuf(4096);
  unsigned short* wv_p  = bfbuf(4096);
  unsigned short* wt_p  = bfbuf(4096);
  unsigned short* wq1_p = bfbuf(4096);
  unsigned short* wk1_p = bfbuf(4096);
  unsigned short* wv1_p = bfbuf(4096);
  unsigned short* wdown_p = bfbuf(8 * 4096);
  unsigned short* w3t_p = bfbuf(27 * 4096);
  unsigned short* w3a_p = bfbuf(27 * 4096);
  unsigned short* w3b_p = bfbuf(27 * 4096);
  // activations
  float* xd_f  = f32buf(NE);  unsigned short* xd16  = bfbuf(NE);
  unsigned short* q16   = bfbuf(NE);
  unsigned short* ke16  = bfbuf(NE);
  unsigned short* veT16 = bfbuf(NE);
  unsigned short* at16  = bfbuf(NE);
  float* xr_f  = f32buf(NE);
  float* xdA_f = f32buf(NE); unsigned short* xdA16 = bfbuf(NE);
  unsigned short* q1_16 = bfbuf(NE);
  float* kvp_f = f32buf(NE); unsigned short* kv16 = bfbuf(NE);
  unsigned short* k1_16  = bfbuf(NE);
  unsigned short* v1T16  = bfbuf(NE);
  unsigned short* o16    = bfbuf(NE);
  float* xr2_f = f32buf(NE);
  float* xdB_f = f32buf(NE);
  unsigned short* z0_16 = bfbuf(NE);
  float* z1_f  = f32buf(NE); unsigned short* z1r16 = bfbuf(NE);
  float* z2_f  = f32buf(NE);
  float* mv    = f32buf(2 * NFEAT);

  const int EB = 256;
  auto cvt = [&](const float* src, unsigned short* dst, int n) {
    cvt_bf16_kernel<<<(n + EB - 1) / EB, EB, 0, stream>>>(src, dst, n);
  };
  auto packw = [&](const float* src, unsigned short* dst, int taps) {
    pack_w_kernel<<<(taps * 4096 + EB - 1) / EB, EB, 0, stream>>>(src, dst, taps);
  };
  cvt(x_decoder, Xd16, NE);
  cvt(x_encoder, Xe16, NE);
  packw(Wp1, wp1_p, 1); packw(Wq, wq_p, 1);   packw(Wk, wk_p, 1);
  packw(Wv, wv_p, 1);   packw(Wt, wt_p, 1);   packw(Wq1, wq1_p, 1);
  packw(Wk1, wk1_p, 1); packw(Wv1, wv1_p, 1);
  packw(Wdown, wdown_p, 8);
  packw(W3t, w3t_p, 27); packw(W3a, w3a_p, 27); packw(W3b, w3b_p, 27);

  const int GB = NDPTS / 64;  // 192 blocks x 128 threads (4 waves)
  const int EG = (NE + EB - 1) / EB;

  // xd = x_decoder @ Wp1 ; q/ke/ve projections (ve written transposed)
  gemm64_kernel<<<GB, 128, 0, stream>>>(Xd16, wp1_p, xd_f, xd16, nullptr, NDPTS);
  gemm64_kernel<<<GB, 128, 0, stream>>>(xd16, wq_p, nullptr, q16, nullptr, NDPTS);
  gemm64_kernel<<<GB, 128, 0, stream>>>(Xe16, wk_p, nullptr, ke16, nullptr, NDPTS);
  gemm64_kernel<<<GB, 128, 0, stream>>>(Xe16, wv_p, nullptr, nullptr, veT16, NDPTS);

  // Stage-1 global flash attention; then xr = attn_out @ Wt
  attn_stage1_kernel<<<GB, 128, 0, stream>>>(q16, ke16, veT16, at16);
  gemm64_kernel<<<GB, 128, 0, stream>>>(at16, wt_p, xr_f, nullptr, nullptr, NDPTS);

  // xd = xd + bn0(xr)
  bn_stats_kernel<<<NFEAT, 256, 0, stream>>>(xr_f, nullptr, NDPTS, mv);
  bn_apply_kernel<<<EG, EB, 0, stream>>>(xr_f, nullptr, xd_f, mv, bn_g + 0 * NFEAT,
                                         bn_b + 0 * NFEAT, 0, xdA_f, xdA16, NE);

  // q1; kv = bn1(gconv(q1, kv_nbr, Wdown)); k1 / v1 (v1 transposed)
  gemm64_kernel<<<GB, 128, 0, stream>>>(xdA16, wq1_p, nullptr, q1_16, nullptr, NDPTS);
  gconv_kernel<<<GB, 128, 0, stream>>>(q1_16, kv_nbr, 8, wdown_p, kvp_f, nullptr);
  bn_stats_kernel<<<NFEAT, 256, 0, stream>>>(kvp_f, nullptr, NDPTS, mv);
  bn_apply_kernel<<<EG, EB, 0, stream>>>(kvp_f, nullptr, nullptr, mv, bn_g + 1 * NFEAT,
                                         bn_b + 1 * NFEAT, 0, nullptr, kv16, NE);
  gemm64_kernel<<<GB, 128, 0, stream>>>(kv16, wk1_p, nullptr, k1_16, nullptr, NDPTS);
  gemm64_kernel<<<GB, 128, 0, stream>>>(kv16, wv1_p, nullptr, nullptr, v1T16, NDPTS);

  // Stage-2 ragged per-batch attention (195 tiles for the fixed counts)
  attn_stage2_kernel<<<195, 128, 0, stream>>>(q1_16, k1_16, v1T16, o16);

  // xr2 = gconv(o, nbr, W3t); xd = xd + bn2(xr2)
  gconv_kernel<<<GB, 128, 0, stream>>>(o16, nbr, 27, w3t_p, xr2_f, nullptr);
  bn_stats_kernel<<<NFEAT, 256, 0, stream>>>(xr2_f, nullptr, NDPTS, mv);
  bn_apply_kernel<<<EG, EB, 0, stream>>>(xr2_f, nullptr, xdA_f, mv, bn_g + 2 * NFEAT,
                                         bn_b + 2 * NFEAT, 0, xdB_f, nullptr, NE);

  // res block: z = conv3(relu(bn3(xd))); z = conv3(relu(bn4(z))); out = relu(bn5(xd+z))
  bn_stats_kernel<<<NFEAT, 256, 0, stream>>>(xdB_f, nullptr, NDPTS, mv);
  bn_apply_kernel<<<EG, EB, 0, stream>>>(xdB_f, nullptr, nullptr, mv, bn_g + 3 * NFEAT,
                                         bn_b + 3 * NFEAT, 1, nullptr, z0_16, NE);
  gconv_kernel<<<GB, 128, 0, stream>>>(z0_16, nbr, 27, w3a_p, z1_f, nullptr);
  bn_stats_kernel<<<NFEAT, 256, 0, stream>>>(z1_f, nullptr, NDPTS, mv);
  bn_apply_kernel<<<EG, EB, 0, stream>>>(z1_f, nullptr, nullptr, mv, bn_g + 4 * NFEAT,
                                         bn_b + 4 * NFEAT, 1, nullptr, z1r16, NE);
  gconv_kernel<<<GB, 128, 0, stream>>>(z1r16, nbr, 27, w3b_p, z2_f, nullptr);
  bn_stats_kernel<<<NFEAT, 256, 0, stream>>>(xdB_f, z2_f, NDPTS, mv);
  bn_apply_kernel<<<EG, EB, 0, stream>>>(xdB_f, z2_f, nullptr, mv, bn_g + 5 * NFEAT,
                                         bn_b + 5 * NFEAT, 1, out, nullptr, NE);
}
